// Attention_24996709663241
// MI455X (gfx1250) — matrix operand order repaired and checked by reading, not compiled, not run on hardware
//
#include <hip/hip_runtime.h>
#include <hip/hip_bf16.h>

typedef __attribute__((ext_vector_type(16))) _Float16 v16h;
typedef __attribute__((ext_vector_type(8)))  float    v8f;

#define BATCH   128
#define NPIX    196
#define DIMC    384
#define SCALEF  0.17677669529663687f   // 32^-0.5
#define BN_EPS  1e-5f

// ---- WMMA fragment packing (this code's own contiguous k order, wave32) -----------------------------
// A 16x32 f16: lane = row&15 + 16*(k>>4);   within-lane pos = k&15
// B 32x16 f16: lane = col    + 16*(k>>4);     within-lane pos = k&15
// Each lane's 16 operand halves are contiguous (32B) -> ds_load_b128 x2.

__device__ __forceinline__ int a_pack_idx(int r, int k) {
  int lane = (r & 15) + ((k >> 4) << 4);
  int pos  = k & 15;
  return lane * 16 + pos;
}
__device__ __forceinline__ int b_pack_idx(int k, int c) {
  int lane = (c & 15) + ((k >> 4) << 4);
  int pos  = k & 15;
  return lane * 16 + pos;
}
__device__ __forceinline__ v16h frag_ld(const _Float16* base, int tile, int lane) {
  return *(const v16h*)(base + tile * 512 + lane * 16);
}
__device__ __forceinline__ v8f wmma16(v16h a, v16h b, v8f c) {
  return __builtin_amdgcn_wmma_f32_16x16x32_f16(false, a, false, b, (short)0, c, false, false);
}

// Async 16B global->LDS copy (per-lane addresses), tracked on ASYNCcnt.
__device__ __forceinline__ void async_b128(const void* gsrc, void* ldst) {
  unsigned           lds  = (unsigned)(size_t)ldst;          // LDS address: low 32 bits of the pointer
  unsigned long long gaddr = (unsigned long long)(size_t)gsrc;
  asm volatile("global_load_async_to_lds_b128 %0, %1, off"
               :: "v"(lds), "v"(gaddr) : "memory");
}
__device__ __forceinline__ void wait_async0() {
  asm volatile("s_wait_asynccnt 0x0" ::: "memory");
}

// ---------------- Kernel 1: QKV GEMM (1536 x 384) x (384 x B*196) + BN -------

__global__ __launch_bounds__(256) void k_qkv(
    const float* __restrict__ x, const float* __restrict__ w,
    const float* __restrict__ gg, const float* __restrict__ bb,
    const float* __restrict__ mm, const float* __restrict__ vv,
    float* __restrict__ qbuf, _Float16* __restrict__ kbuf, _Float16* __restrict__ vbuf)
{
  __shared__ __align__(32) _Float16 Apk[8 * 512];
  __shared__ __align__(32) _Float16 Bpk[4 * 512];
  const int tid  = threadIdx.x;
  const int wave = tid >> 5;
  const int lane = tid & 31;
  const int l = lane & 15, hi = lane >> 4;
  const int oBase = blockIdx.x * 128;
  const int nBase = blockIdx.y * 64;
  const int b = blockIdx.z;

  v8f acc[4] = {};
  for (int kb = 0; kb < DIMC; kb += 32) {
    __syncthreads();
    for (int i = tid; i < 128 * 32; i += 256) {
      int r = i >> 5, k = i & 31;
      Apk[(r >> 4) * 512 + a_pack_idx(r, k)] =
          (_Float16)w[(size_t)(oBase + r) * DIMC + kb + k];
    }
    for (int i = tid; i < 32 * 64; i += 256) {
      int k = i >> 6, c = i & 63;
      int n = nBase + c;
      float xv = (n < NPIX) ? x[((size_t)b * DIMC + kb + k) * NPIX + n] : 0.f;
      Bpk[(c >> 4) * 512 + b_pack_idx(k, c & 15)] = (_Float16)xv;
    }
    __syncthreads();
    if (kb + 32 < DIMC)
      __builtin_prefetch(&x[((size_t)b * DIMC + kb + 32 + (tid & 31)) * NPIX + nBase], 0, 1);

    v16h a = frag_ld(Apk, wave, lane);
    v16h bf[4];
#pragma unroll
    for (int ct = 0; ct < 4; ++ct) bf[ct] = frag_ld(Bpk, ct, lane);
#pragma unroll
    for (int ct = 0; ct < 4; ++ct) acc[ct] = wmma16(a, bf[ct], acc[ct]);
  }

#pragma unroll
  for (int ct = 0; ct < 4; ++ct) {
    int n = nBase + ct * 16 + l;
    if (n >= NPIX) continue;
#pragma unroll
    for (int r = 0; r < 8; ++r) {
      int o = oBase + wave * 16 + r + 8 * hi;
      float s = gg[o] * rsqrtf(vv[o] + BN_EPS);
      float t = bb[o] - mm[o] * s;
      float val = acc[ct][r] * s + t;
      if (o < 256) {
        qbuf[((size_t)b * 256 + o) * NPIX + n] = val;
      } else if (o < 512) {
        int c = o - 256, h = c >> 5, d = c & 31;
        kbuf[(((size_t)b * 8 + h) * 32 + d) * NPIX + n] = (_Float16)val;
      } else {
        int c = o - 512, h = c >> 7, d = c & 127;
        vbuf[(((size_t)b * 8 + h) * NPIX + n) * 128 + d] = (_Float16)val;
      }
    }
  }
}

// ---------------- Kernel 2: 3x3 depthwise conv on q + BN -> q^T f16 ----------

__global__ __launch_bounds__(256) void k_dwconv(
    const float* __restrict__ qbuf, const float* __restrict__ wdw,
    const float* __restrict__ gg, const float* __restrict__ bb,
    const float* __restrict__ mm, const float* __restrict__ vv,
    _Float16* __restrict__ qT)
{
  int idx = blockIdx.x * 256 + threadIdx.x;
  const int total = BATCH * 256 * NPIX;
  if (idx >= total) return;
  int n  = idx % NPIX;
  int ch = (idx / NPIX) % 256;
  int b  = idx / (NPIX * 256);
  int py = n / 14, px = n % 14;
  const float* src = qbuf + ((size_t)b * 256 + ch) * NPIX;
  float a = 0.f;
#pragma unroll
  for (int dy = -1; dy <= 1; ++dy) {
    int yy = py + dy;
    if (yy < 0 || yy >= 14) continue;
#pragma unroll
    for (int dx = -1; dx <= 1; ++dx) {
      int xx = px + dx;
      if (xx < 0 || xx >= 14) continue;
      a += src[yy * 14 + xx] * wdw[ch * 9 + (dy + 1) * 3 + (dx + 1)];
    }
  }
  float s = gg[ch] * rsqrtf(vv[ch] + BN_EPS);
  float t = bb[ch] - mm[ch] * s;
  float val = a * s + t;
  int h = ch >> 5, d = ch & 31;
  qT[(((size_t)b * 8 + h) * NPIX + n) * 32 + d] = (_Float16)val;
}

// ---------------- Kernel 3: attention per (b,h) ------------------------------
// S = q^T k * scale + ab[h, idxs];  P = softmax(S);  O = relu(P v^T)

__global__ __launch_bounds__(256) void k_attn(
    const _Float16* __restrict__ qT, const _Float16* __restrict__ kbuf,
    const _Float16* __restrict__ vbuf,
    const float* __restrict__ ab, const int* __restrict__ idxs,
    _Float16* __restrict__ act)
{
  extern __shared__ char smem[];
  _Float16* qpk = (_Float16*)smem;             // 16 row-tiles (256 rows) packed A
  _Float16* kpk = qpk + 16 * 512;              // 13 col-tiles packed B (K=32)
  _Float16* vpk = kpk + 13 * 512;              // [7 kb][8 ct] packed B
  float*    Sb  = (float*)(vpk + 56 * 512);    // [64][208] f32
  _Float16* Ppk = (_Float16*)(Sb + 64 * 208);  // [4 rt][7 kb] packed A

  const int tid  = threadIdx.x;
  const int wave = tid >> 5, lane = tid & 31;
  const int l = lane & 15, hi = lane >> 4;
  const int bh = blockIdx.x;
  const int b = bh >> 3, h = bh & 7;

  const _Float16* qg = qT   + (size_t)bh * NPIX * 32;
  const _Float16* kg = kbuf + (size_t)bh * 32 * NPIX;
  const _Float16* vg = vbuf + (size_t)bh * NPIX * 128;

  // q^T: async 16B chunks (each lane's packed 8-half group is contiguous in
  // global: row n, d-octet j -> lane (n&15)+16*(j>>1), pos 8*(j&1))
  for (int i = tid; i < 196 * 4; i += 256) {
    int n = i >> 2, j = i & 3;
    int dstIdx = (n >> 4) * 512 + ((n & 15) + ((j >> 1) << 4)) * 16 + ((j & 1) << 3);
    async_b128(qg + n * 32 + j * 8, qpk + dstIdx);
  }
  for (int i = 196 * 32 + tid; i < 256 * 32; i += 256) {   // zero pad rows 196..255
    int n = i >> 5, d = i & 31;
    qpk[(n >> 4) * 512 + a_pack_idx(n, d)] = (_Float16)0.f;
  }
  for (int i = tid; i < 32 * 208; i += 256) {
    int d = i / 208, m = i % 208;
    _Float16 val = (m < NPIX) ? kg[d * NPIX + m] : (_Float16)0.f;
    kpk[(m >> 4) * 512 + b_pack_idx(d, m & 15)] = val;
  }
  for (int i = tid; i < 224 * 128; i += 256) {
    int m = i >> 7, d = i & 127;
    _Float16 val = (m < NPIX) ? vg[m * 128 + d] : (_Float16)0.f;
    vpk[((m >> 5) * 8 + (d >> 4)) * 512 + b_pack_idx(m & 31, d & 15)] = val;
  }
  wait_async0();
  __syncthreads();

  for (int rb = 0; rb < 4; ++rb) {
    const int rowBase = rb * 64;

    // Phase A: S block (64 x 196); KEY_DIM=32 => one WMMA per 16x16 tile
    for (int t = wave; t < 52; t += 8) {
      int rt = t / 13, ctile = t % 13;
      v16h a   = frag_ld(qpk, rb * 4 + rt, lane);
      v16h bfr = frag_ld(kpk, ctile, lane);
      v8f accv = {};
      accv = wmma16(a, bfr, accv);
      int mcol = ctile * 16 + l;
#pragma unroll
      for (int r = 0; r < 8; ++r) {
        int lr = rt * 16 + r + 8 * hi;
        int n  = rowBase + lr;
        float valw = 0.f;
        if (n < NPIX && mcol < NPIX) {
          int iv = idxs[n * NPIX + mcol];
          valw = accv[r] * SCALEF + ab[h * NPIX + iv];
        }
        Sb[lr * 208 + mcol] = valw;
      }
    }
    __syncthreads();

    // Phase B: row softmax; write P directly in packed-A layout (K padded->224)
    if (tid < 64) {
      int lr = tid;
      int n  = rowBase + lr;
      int rt = lr >> 4;
      if (n < NPIX) {
        float mx = -1e30f;
        for (int m = 0; m < NPIX; ++m) mx = fmaxf(mx, Sb[lr * 208 + m]);
        float sum = 0.f;
        for (int m = 0; m < NPIX; ++m) {
          float e = __expf(Sb[lr * 208 + m] - mx);
          Sb[lr * 208 + m] = e;
          sum += e;
        }
        float inv = 1.f / sum;
        for (int m = 0; m < 224; ++m) {
          float p = (m < NPIX) ? Sb[lr * 208 + m] * inv : 0.f;
          Ppk[(rt * 7 + (m >> 5)) * 512 + a_pack_idx(lr, m & 31)] = (_Float16)p;
        }
      } else {
        for (int m = 0; m < 224; ++m)
          Ppk[(rt * 7 + (m >> 5)) * 512 + a_pack_idx(lr, m & 31)] = (_Float16)0.f;
      }
    }
    __syncthreads();

    // Phase C: O block = P(64x224) * vT(224x128), relu, store (bn, c) f16
    {
      int rt   = wave >> 1;
      int cgrp = (wave & 1) * 4;
      v8f acco[4] = {};
#pragma unroll
      for (int kt = 0; kt < 7; ++kt) {
        v16h a = frag_ld(Ppk, rt * 7 + kt, lane);
        v16h bfr[4];
#pragma unroll
        for (int ct = 0; ct < 4; ++ct) bfr[ct] = frag_ld(vpk, kt * 8 + cgrp + ct, lane);
#pragma unroll
        for (int ct = 0; ct < 4; ++ct) acco[ct] = wmma16(a, bfr[ct], acco[ct]);
      }
#pragma unroll
      for (int ct = 0; ct < 4; ++ct) {
        int d = (cgrp + ct) * 16 + l;
#pragma unroll
        for (int r = 0; r < 8; ++r) {
          int n = rowBase + rt * 16 + r + 8 * hi;
          if (n < NPIX) {
            float vo = fmaxf(acco[ct][r], 0.f);
            act[((size_t)b * NPIX + n) * 1024 + h * 128 + d] = (_Float16)vo;
          }
        }
      }
    }
    __syncthreads();
  }
}

// ---------------- Kernel 4: proj GEMM (384 x 1024) x (1024 x B*196) + BN -----

__global__ __launch_bounds__(256) void k_proj(
    const _Float16* __restrict__ act, const float* __restrict__ wp,
    const float* __restrict__ gg, const float* __restrict__ bb,
    const float* __restrict__ mm, const float* __restrict__ vv,
    float* __restrict__ out)
{
  __shared__ __align__(32) _Float16 Apk[8 * 512];
  __shared__ __align__(32) _Float16 Bpk[4 * 512];
  const int tid  = threadIdx.x;
  const int wave = tid >> 5, lane = tid & 31;
  const int l = lane & 15, hi = lane >> 4;
  const int oBase = blockIdx.x * 128;
  const int nBase = blockIdx.y * 64;
  const int b = blockIdx.z;

  v8f acc[4] = {};
  for (int kb = 0; kb < 1024; kb += 32) {
    __syncthreads();
    for (int i = tid; i < 128 * 32; i += 256) {
      int r = i >> 5, k = i & 31;
      Apk[(r >> 4) * 512 + a_pack_idx(r, k)] =
          (_Float16)wp[(size_t)(oBase + r) * 1024 + kb + k];
    }
    // B stage: act f16 (bn,c) -> packed B. 16 consecutive K land in one lane's
    // contiguous 32B region: use async global->LDS b128 x2 per (col, K-half).
    for (int i = tid; i < 128; i += 256) {
      int c = i >> 1, half = i & 1;
      int n = nBase + c;
      _Float16* dst = Bpk + (c >> 4) * 512 + ((c & 15) + (half << 4)) * 16;
      if (n < NPIX) {
        const _Float16* src = act + ((size_t)b * NPIX + n) * 1024 + kb + half * 16;
        async_b128(src, dst);
        async_b128(src + 8, dst + 8);
      } else {
        v16h z = {};
        *(v16h*)dst = z;
      }
    }
    wait_async0();
    __syncthreads();
    if (kb + 32 < 1024 && nBase < NPIX)
      __builtin_prefetch(&act[((size_t)b * NPIX + nBase) * 1024 + kb + 32], 0, 1);

    v16h a = frag_ld(Apk, wave, lane);
    v16h bf[4];
#pragma unroll
    for (int ct = 0; ct < 4; ++ct) bf[ct] = frag_ld(Bpk, ct, lane);
#pragma unroll
    for (int ct = 0; ct < 4; ++ct) acc[ct] = wmma16(a, bf[ct], acc[ct]);
  }

#pragma unroll
  for (int ct = 0; ct < 4; ++ct) {
    int n = nBase + ct * 16 + l;
    if (n >= NPIX) continue;
#pragma unroll
    for (int r = 0; r < 8; ++r) {
      int o = oBase + wave * 16 + r + 8 * hi;
      float s = gg[o] * rsqrtf(vv[o] + BN_EPS);
      float t = bb[o] - mm[o] * s;
      out[((size_t)b * 384 + o) * NPIX + n] = acc[ct][r] * s + t;
    }
  }
}

// ---------------- Host launch ------------------------------------------------

extern "C" void kernel_launch(void* const* d_in, const int* in_sizes, int n_in,
                              void* d_out, int out_size, void* d_ws, size_t ws_size,
                              hipStream_t stream)
{
  (void)in_sizes; (void)n_in; (void)out_size; (void)ws_size;
  const float* x      = (const float*)d_in[0];
  const float* w_qkv  = (const float*)d_in[1];
  const float* g_qkv  = (const float*)d_in[2];
  const float* b_qkv  = (const float*)d_in[3];
  const float* m_qkv  = (const float*)d_in[4];
  const float* v_qkv  = (const float*)d_in[5];
  const float* w_dw   = (const float*)d_in[6];
  const float* g_dw   = (const float*)d_in[7];
  const float* b_dw   = (const float*)d_in[8];
  const float* m_dw   = (const float*)d_in[9];
  const float* v_dw   = (const float*)d_in[10];
  const float* w_proj = (const float*)d_in[11];
  const float* g_proj = (const float*)d_in[12];
  const float* b_proj = (const float*)d_in[13];
  const float* m_proj = (const float*)d_in[14];
  const float* v_proj = (const float*)d_in[15];
  const float* ab     = (const float*)d_in[16];
  const int*   idxs   = (const int*)d_in[17];
  float* out = (float*)d_out;

  char* ws = (char*)d_ws;
  float*    qbuf = (float*)ws;    ws += (size_t)BATCH * 256  * NPIX * sizeof(float);
  _Float16* kbuf = (_Float16*)ws; ws += (size_t)BATCH * 256  * NPIX * sizeof(_Float16);
  _Float16* vbuf = (_Float16*)ws; ws += (size_t)BATCH * 1024 * NPIX * sizeof(_Float16);
  _Float16* qTb  = (_Float16*)ws; ws += (size_t)BATCH * 256  * NPIX * sizeof(_Float16);
  _Float16* actb = (_Float16*)ws; ws += (size_t)BATCH * 1024 * NPIX * sizeof(_Float16);

  k_qkv<<<dim3(12, 4, BATCH), 256, 0, stream>>>(
      x, w_qkv, g_qkv, b_qkv, m_qkv, v_qkv, qbuf, kbuf, vbuf);

  int tot = BATCH * 256 * NPIX;
  k_dwconv<<<(tot + 255) / 256, 256, 0, stream>>>(
      qbuf, w_dw, g_dw, b_dw, m_dw, v_dw, qTb);

  size_t smemBytes = (size_t)(16 * 512 + 13 * 512 + 56 * 512) * sizeof(_Float16)
                   + (size_t)64 * 208 * sizeof(float)
                   + (size_t)28 * 512 * sizeof(_Float16);   // 168960 B total
  k_attn<<<BATCH * 8, 256, smemBytes, stream>>>(qTb, kbuf, vbuf, ab, idxs, actb);

  k_proj<<<dim3(3, 4, BATCH), 256, 0, stream>>>(
      actb, w_proj, g_proj, b_proj, m_proj, v_proj, out);
}
